// AttnDecoderLSTM_39453569581311
// MI455X (gfx1250) — matrix operand order repaired and checked by reading, not compiled, not run on hardware
//
#include <hip/hip_runtime.h>
#include <math.h>

// ---- problem constants (reference: B=32, T=128, H=1024, V=32000, SOS=1) ----
constexpr int kB  = 32;
constexpr int kT  = 128;
constexpr int kH  = 1024;
constexpr int kV  = 32000;
constexpr int kG  = 4 * kH;   // 4096 gate columns
constexpr int kK2 = 2 * kH;   // 2048 concat-K  (h | emb)
constexpr int kNWG = 32;      // workgroups in persistent recurrence kernel
constexpr int kHB = kB * kH;  // one h slice (32 x 1024)

typedef __attribute__((ext_vector_type(16))) __bf16 v16bf;
typedef __attribute__((ext_vector_type(8)))  __bf16 v8bf;
typedef __attribute__((ext_vector_type(8)))  float  v8f;

__device__ __forceinline__ __bf16 f2bf(float x) { return (__bf16)x; }

__device__ __forceinline__ v8f wmma_bf16(v16bf a, v16bf b, v8f c) {
  // D = A(16x32 bf16) * B(32x16 bf16) + C(16x16 f32)
  return __builtin_amdgcn_wmma_f32_16x16x32_bf16(
      false, a, false, b, (short)0, c, false, false);
}

// A fragment (16x32 bf16): lane m = lane&15, K = kh*8..+7 and 16+kh*8..+7
__device__ __forceinline__ v16bf load_a_bf(const __bf16* base, int mrow, int kb0) {
  const __bf16* p = base + (size_t)mrow * kH + kb0;
  v8bf lo = *(const v8bf*)p;
  v8bf hi = *(const v8bf*)(p + 16);
  v16bf av;
#pragma unroll
  for (int e = 0; e < 8; ++e) { av[e] = lo[e]; av[e + 8] = hi[e]; }
  return av;
}
// B fragment (32x16 from row-major W[n][k]): lane n = lane&15, K = kh*8..+7 and 16+kh*8..+7
__device__ __forceinline__ v16bf load_b(const __bf16* __restrict__ W,
                                        int n, int kb, int ldk) {
  const __bf16* p = W + (size_t)n * ldk + (kb - (kb & 16) + ((kb & 16) >> 1));
  v8bf lo = *(const v8bf*)p;
  v8bf hi = *(const v8bf*)(p + 16);
  v16bf bv;
#pragma unroll
  for (int e = 0; e < 8; ++e) { bv[e] = lo[e]; bv[e + 8] = hi[e]; }
  return bv;
}

// ---------------- prep kernels ----------------

__global__ void k_prep_wcat(const float* __restrict__ Whh,
                            const float* __restrict__ Wih,
                            __bf16* __restrict__ Wcat) {
  size_t i = (size_t)blockIdx.x * blockDim.x + threadIdx.x;  // 4096*2048
  int k = (int)(i & (kK2 - 1));
  int n = (int)(i >> 11);
  float v = (k < kH) ? Whh[(size_t)n * kH + k]
                     : Wih[(size_t)n * (3 * kH) + kH + (k - kH)];
  Wcat[i] = f2bf(v);
}

__global__ void k_prep_emb(const float* __restrict__ E,
                           const int* __restrict__ tgt,
                           __bf16* __restrict__ Aemb) {
  size_t i = (size_t)blockIdx.x * blockDim.x + threadIdx.x;  // 128*32*1024
  int k = (int)(i & (kH - 1));
  int b = (int)((i >> 10) & 31);
  int t = (int)(i >> 15);
  int tok = (t == 0) ? 1 : tgt[b * kT + (t - 1)];
  Aemb[i] = f2bf(E[(size_t)tok * kH + k]);
}

__global__ void k_prep_wout(const float* __restrict__ W, __bf16* __restrict__ Wb,
                            size_t n) {
  size_t i = (size_t)blockIdx.x * blockDim.x + threadIdx.x;
  if (i < n) Wb[i] = f2bf(W[i]);
}

// g_enc[b][n] = b_ih[n] + b_hh[n] + sum_k enc[b][k]*(W_ih[n][k] + W_ih[n][2H+k])
__global__ void k_genc(const float* __restrict__ enc, const float* __restrict__ Wih,
                       const float* __restrict__ bih, const float* __restrict__ bhh,
                       float* __restrict__ genc) {
  int tid = blockIdx.x * blockDim.x + threadIdx.x;  // 32*4096
  int b = tid & 31;
  int n = tid >> 5;
  const float* er = enc + (size_t)b * kH;
  const float* wr = Wih + (size_t)n * (3 * kH);
  float s = bih[n] + bhh[n];
  for (int k = 0; k < kH; ++k) s += er[k] * (wr[k] + wr[2 * kH + k]);
  genc[(size_t)b * kG + n] = s;
}

// h state: Hseq slice 0 = bf16(enc); f32 scratch h; c = enc; barrier = 0
__global__ void k_init(const float* __restrict__ enc, __bf16* __restrict__ Hseq,
                       float* __restrict__ hF, float* __restrict__ cst,
                       unsigned* __restrict__ bar) {
  int i = blockIdx.x * blockDim.x + threadIdx.x;  // 32*1024
  float v = enc[i];
  Hseq[i] = f2bf(v);
  hF[i] = v;
  cst[i] = v;
  if (i == 0) bar[0] = 0u;
}

// ---------------- persistent recurrence kernel ----------------
// Monotonic grid barrier: release add + acquire spin (agent scope -> L2 coherent,
// acquire invalidates WGP$ so next step's h loads are fresh).
__device__ __forceinline__ void grid_sync(unsigned* bar, unsigned target) {
  __syncthreads();
  if (threadIdx.x == 0) {
    __hip_atomic_fetch_add(bar, 1u, __ATOMIC_RELEASE, __HIP_MEMORY_SCOPE_AGENT);
    while (__hip_atomic_load(bar, __ATOMIC_ACQUIRE, __HIP_MEMORY_SCOPE_AGENT) <
           target) {
      __builtin_amdgcn_s_sleep(1);
    }
  }
  __syncthreads();
}

// 32 workgroups; wg owns gate-columns [wg*32, wg*32+32) of all 4 gates
// -> 512 KB weight slab per WGP per step streaming from L2 (the step bottleneck).
// Wave w: gate = w>>1, N-tile = w&1. Per step: M=32, N=128, K=2048 GEMM.
// Step t reads bf16 h from Hseq slice t and writes slice t+1 (no f32 restage).
__global__ void __launch_bounds__(256) k_steps_all(
    const __bf16* __restrict__ Wcat, const __bf16* __restrict__ Aemb,
    const float* __restrict__ genc, __bf16* __restrict__ Hseq,
    float* __restrict__ hF, float* __restrict__ cst,
    unsigned* __restrict__ bar) {
  __shared__ float gbuf[4 * 32 * 32];  // [gate][m][jlocal], 16 KB

  const int j0   = blockIdx.x * 32;
  const int lane = threadIdx.x & 31;
  const int wv   = threadIdx.x >> 5;
  const int gate = wv >> 1;
  const int tt   = wv & 1;
  const int mlo  = lane & 15;
  const int kh   = lane >> 4;
  const int n0   = gate * kH + j0 + tt * 16 + mlo;  // this wave's B column

  for (int t = 0; t < kT; ++t) {
    const __bf16* Hprev = Hseq + (size_t)t * kHB;        // h_t  (bf16, 64 KB)
    __bf16*       Hnext = Hseq + (size_t)(t + 1) * kHB;  // h_{t+1}
    const __bf16* Ae    = Aemb + (size_t)t * kHB;        // emb_t

    v8f acc[2] = {};

    // ---- phase 1: K-tiles 0..31, A = previous h slice (global bf16) ----
    {
      v16bf a0[2], a1[2], b0[2];
      a0[0] = load_a_bf(Hprev, mlo, kh * 8);
      a1[0] = load_a_bf(Hprev, 16 + mlo, kh * 8);
      b0[0] = load_b(Wcat, n0, kh * 16, kK2);
#pragma unroll 2
      for (int kt = 0; kt < 32; ++kt) {
        const int cur = kt & 1, nxt = cur ^ 1;
        const int kp = (kt + 1) * 32;  // unmasked: last prefetch dead, in-pad
        a0[nxt] = load_a_bf(Hprev, mlo, kp + kh * 8);
        a1[nxt] = load_a_bf(Hprev, 16 + mlo, kp + kh * 8);
        b0[nxt] = load_b(Wcat, n0, kp + kh * 16, kK2);
        acc[0] = wmma_bf16(a0[cur], b0[cur], acc[0]);
        acc[1] = wmma_bf16(a1[cur], b0[cur], acc[1]);
      }
    }
    // ---- phase 2: K-tiles 32..63, A = pre-gathered bf16 embeddings ----
    {
      v16bf a0[2], a1[2], b0[2];
      a0[0] = load_a_bf(Ae, mlo, kh * 8);
      a1[0] = load_a_bf(Ae, 16 + mlo, kh * 8);
      b0[0] = load_b(Wcat, n0, kH + kh * 16, kK2);
#pragma unroll 2
      for (int kt = 0; kt < 32; ++kt) {
        const int cur = kt & 1, nxt = cur ^ 1;
        const int kp = (kt + 1) * 32;
        a0[nxt] = load_a_bf(Ae, mlo, kp + kh * 8);
        a1[nxt] = load_a_bf(Ae, 16 + mlo, kp + kh * 8);
        b0[nxt] = load_b(Wcat, n0, kH + kp + kh * 16, kK2);
        acc[0] = wmma_bf16(a0[cur], b0[cur], acc[0]);
        acc[1] = wmma_bf16(a1[cur], b0[cur], acc[1]);
      }
    }

    // C layout: VGPR r -> m = r + 8*kh, n = tile*16 + (lane&15). Stage gates.
    {
      const int jl   = tt * 16 + mlo;          // 0..31 within this wg's columns
      const int ncol = gate * kH + j0 + jl;
#pragma unroll
      for (int mt = 0; mt < 2; ++mt)
#pragma unroll
        for (int r = 0; r < 8; ++r) {
          const int m = mt * 16 + kh * 8 + r;
          gbuf[(gate * 32 + m) * 32 + jl] =
              acc[mt][r] + genc[(size_t)m * kG + ncol];
        }
    }
    __syncthreads();

    // LSTM pointwise for this workgroup's 32 columns (c is WG-private)
    for (int i = threadIdx.x; i < 32 * 32; i += 256) {
      const int m = i >> 5, jl = i & 31;
      const int col = j0 + jl;
      const float ig = gbuf[(0 * 32 + m) * 32 + jl];
      const float fg = gbuf[(1 * 32 + m) * 32 + jl];
      const float gg = gbuf[(2 * 32 + m) * 32 + jl];
      const float og = gbuf[(3 * 32 + m) * 32 + jl];
      const float cp = cst[(size_t)m * kH + col];
      const float si = 1.f / (1.f + __expf(-ig));
      const float sf = 1.f / (1.f + __expf(-fg));
      const float so = 1.f / (1.f + __expf(-og));
      const float c2 = sf * cp + si * tanhf(gg);
      const float h2 = so * tanhf(c2);
      cst[(size_t)m * kH + col] = c2;
      hF[(size_t)m * kH + col] = h2;               // f32; only final step used
      Hnext[(size_t)m * kH + col] = f2bf(h2);      // feeds step t+1 and k_logits
    }

    grid_sync(bar, (unsigned)(t + 1) * kNWG);  // also orders gbuf reuse
  }
}

// ---------------- batched output projection: [4096 x 1024] @ [1024 x 32000] ----
// grid = (V/256, 4096/64); workgroup tile M=64 x N=256; wave = 4 M-tiles x 2 N-tiles
// (B fragments amortized over 4 M-tiles; W_out bf16 copy is L2-resident).
__global__ void __launch_bounds__(256) k_logits(
    const __bf16* __restrict__ Hall, const __bf16* __restrict__ Wout,
    const float* __restrict__ bout, float* __restrict__ out) {
  const int nb    = blockIdx.x;
  const int mbase = blockIdx.y * 64;
  const int wv    = threadIdx.x >> 5;
  const int lane  = threadIdx.x & 31;
  const int mlo   = lane & 15, kh = lane >> 4;
  const int n0    = nb * 256 + (wv * 2) * 16 + mlo;
  const int n1    = n0 + 16;

  v8f acc[4][2] = {};
  v16bf a[2][4], b0[2], b1[2];
#pragma unroll
  for (int mt = 0; mt < 4; ++mt)
    a[0][mt] = load_a_bf(Hall, mbase + mt * 16 + mlo, kh * 8);
  b0[0] = load_b(Wout, n0, kh * 16, kH);
  b1[0] = load_b(Wout, n1, kh * 16, kH);

#pragma unroll 2
  for (int kt = 0; kt < 32; ++kt) {
    const int cur = kt & 1, nxt = cur ^ 1;
    const int kp = (kt + 1) * 32;  // unmasked: last prefetch dead, in-pad
#pragma unroll
    for (int mt = 0; mt < 4; ++mt)
      a[nxt][mt] = load_a_bf(Hall, mbase + mt * 16 + mlo, kp + kh * 8);
    b0[nxt] = load_b(Wout, n0, kp + kh * 16, kH);
    b1[nxt] = load_b(Wout, n1, kp + kh * 16, kH);
#pragma unroll
    for (int mt = 0; mt < 4; ++mt) {
      acc[mt][0] = wmma_bf16(a[cur][mt], b0[cur], acc[mt][0]);
      acc[mt][1] = wmma_bf16(a[cur][mt], b1[cur], acc[mt][1]);
    }
  }

#pragma unroll
  for (int mt = 0; mt < 4; ++mt)
#pragma unroll
    for (int nt = 0; nt < 2; ++nt) {
      const int v = nb * 256 + (wv * 2 + nt) * 16 + mlo;
      const float bo = bout[v];
#pragma unroll
      for (int r = 0; r < 8; ++r) {
        const int row = mbase + mt * 16 + kh * 8 + r;  // row = t*32 + b
        const int tt = row >> 5, bb = row & 31;
        out[((size_t)bb * kT + tt) * kV + v] = acc[mt][nt][r] + bo;  // [B,T,V]
      }
    }
}

// hT, cT, attentions (== 1.0 exactly: softmax over a size-1 axis)
__global__ void k_final(const float* __restrict__ hT, const float* __restrict__ cT,
                        float* __restrict__ out) {
  int i = blockIdx.x * blockDim.x + threadIdx.x;
  size_t base = (size_t)kB * kT * kV;
  if (i < kB * kH)          out[base + i] = hT[i];
  else if (i < 2 * kB * kH) out[base + i] = cT[i - kB * kH];
  else                      out[base + i] = 1.0f;
}

extern "C" void kernel_launch(void* const* d_in, const int* in_sizes, int n_in,
                              void* d_out, int out_size, void* d_ws, size_t ws_size,
                              hipStream_t stream) {
  (void)in_sizes; (void)n_in; (void)out_size; (void)ws_size;
  const float* enc  = (const float*)d_in[0];
  const int*   tgt  = (const int*)d_in[1];
  const float* E    = (const float*)d_in[2];
  // d_in[3..8] (Wa,b_wa,Ua,b_ua,Va,b_va) are mathematically dead:
  // softmax over a size-1 axis is identically 1, so context == enc.
  const float* Wih  = (const float*)d_in[9];
  const float* Whh  = (const float*)d_in[10];
  const float* bih  = (const float*)d_in[11];
  const float* bhh  = (const float*)d_in[12];
  const float* Wout = (const float*)d_in[13];
  const float* bout = (const float*)d_in[14];
  float* out = (float*)d_out;

  // Region order matters: dead tail-prefetches (<= 2 KB overreach) from the
  // unmasked pipelines always land in the next mapped region / tail pad.
  char* ws = (char*)d_ws;
  size_t off = 0;
  __bf16* Wcat  = (__bf16*)(ws + off); off += (size_t)kG * kK2 * 2;        // 16 MB
  __bf16* Aemb  = (__bf16*)(ws + off); off += (size_t)kT * kHB * 2;        // 8 MB
  __bf16* Woutb = (__bf16*)(ws + off); off += (size_t)kV * kH * 2;         // 64 MB
  __bf16* Hseq  = (__bf16*)(ws + off); off += (size_t)(kT + 1) * kHB * 2;  // 8.25 MB
  float*  genc  = (float*)(ws + off);  off += (size_t)kB * kG * 4;
  float*  hF    = (float*)(ws + off);  off += (size_t)kHB * 4;
  float*  cst   = (float*)(ws + off);  off += (size_t)kHB * 4;
  unsigned* bar = (unsigned*)(ws + off); off += 256;
  off += 8192;  // tail pad for dead prefetches

  k_prep_wcat<<<(kG * kK2) / 256, 256, 0, stream>>>(Whh, Wih, Wcat);
  k_prep_emb <<<(kT * kHB) / 256, 256, 0, stream>>>(E, tgt, Aemb);
  k_prep_wout<<<((size_t)kV * kH) / 256, 256, 0, stream>>>(Wout, Woutb,
                                                           (size_t)kV * kH);
  k_genc     <<<(kB * kG) / 256, 256, 0, stream>>>(enc, Wih, bih, bhh, genc);
  k_init     <<<kHB / 256, 256, 0, stream>>>(enc, Hseq, hF, cst, bar);

  // one persistent kernel for all 128 timesteps (grid-wide atomic barrier)
  k_steps_all<<<kNWG, 256, 0, stream>>>(Wcat, Aemb, genc, Hseq, hF, cst, bar);

  // slices 1..128 of Hseq are h_1..h_128 = the per-step outputs
  k_logits<<<dim3(kV / 256, (kT * kB) / 64), 256, 0, stream>>>(
      Hseq + kHB, Woutb, bout, out);
  k_final <<<(2 * kB * kH + kB * kT) / 256, 256, 0, stream>>>(hF, cst, out);
}
